// SimpleUNetBasicBlock_17145509446061
// MI455X (gfx1250) — matrix operand order repaired and checked by reading, not compiled, not run on hardware
//
#include <hip/hip_runtime.h>
#include <hip/hip_bf16.h>

// ---------------------------------------------------------------------------
// Types for CDNA5 WMMA
// ---------------------------------------------------------------------------
typedef __attribute__((ext_vector_type(16))) __bf16 v16bf;
typedef __attribute__((ext_vector_type(8)))  __bf16 v8bf;
typedef __attribute__((ext_vector_type(8)))  float  v8f;

#define CCH 32            // channels (structural constant)
#define BN_EPS 1e-5f
#define TPW 4             // 16-pair tiles processed per wave (B reused)

// round-to-nearest-even f32 -> bf16 bits
__device__ __forceinline__ unsigned short f2bf(float f) {
    unsigned u = __float_as_uint(f);
    unsigned r = u + 0x7FFFu + ((u >> 16) & 1u);
    return (unsigned short)(r >> 16);
}

// ---------------------------------------------------------------------------
// Utility: zero an f32 buffer (grid-stride)
// ---------------------------------------------------------------------------
__global__ void zero_f32(float* __restrict__ p, long long n) {
    long long i = (long long)blockIdx.x * blockDim.x + threadIdx.x;
    long long stride = (long long)gridDim.x * blockDim.x;
    for (; i < n; i += stride) p[i] = 0.0f;
}

// ---------------------------------------------------------------------------
// Convert x (f32 [N,32]) to bf16, vectorized 4-wide
// ---------------------------------------------------------------------------
__global__ void convert_x_bf16(const float* __restrict__ x,
                               unsigned short* __restrict__ xb, long long n4) {
    long long i = (long long)blockIdx.x * blockDim.x + threadIdx.x;
    long long stride = (long long)gridDim.x * blockDim.x;
    for (; i < n4; i += stride) {
        float4 v = ((const float4*)x)[i];
        ushort4 o;
        o.x = f2bf(v.x); o.y = f2bf(v.y); o.z = f2bf(v.z); o.w = f2bf(v.w);
        ((ushort4*)xb)[i] = o;
    }
}

// ---------------------------------------------------------------------------
// Pack W [K,32,32] f32 into per-lane bf16 B-matrix layout.
// For offset k, tile n (cols n*16..n*16+15):
//   lane L holds column (L&15), K-values (L>>4)*8+j (j<8), 16+(L>>4)*8+(j-8) (j>=8)
// Entry index: ((k*2+n)*32 + lane)*16 + j
// ---------------------------------------------------------------------------
__global__ void pack_w_bf16(const float* __restrict__ W,
                            unsigned short* __restrict__ Wlay, int total) {
    int i = blockIdx.x * blockDim.x + threadIdx.x;
    if (i >= total) return;
    int j    = i & 15;
    int lane = (i >> 4) & 31;
    int n    = (i >> 9) & 1;
    int k    = i >> 10;
    int col  = n * 16 + (lane & 15);
    int kk   = (lane >> 4) * 8 + (j & 7) + (j >> 3) * 16;
    Wlay[i] = f2bf(W[(k * CCH + kk) * CCH + col]);
}

// ---------------------------------------------------------------------------
// Sparse conv: one wave32 handles TPW consecutive 16-pair tiles of kernel
// offset k = blockIdx.y.
//  - B tiles (32x16 bf16 each) loaded ONCE per wave, reused TPW times
//  - per tile: gather 16 rows of x_bf16 into the WMMA 16-bit A layout,
//    2x v_wmma_f32_16x16x32_bf16, scatter-add 16x32 f32 via global atomics
// ---------------------------------------------------------------------------
__global__ void __launch_bounds__(256)
sparse_conv_wmma(const unsigned short* __restrict__ xb,
                 const unsigned short* __restrict__ Wlay,
                 const int* __restrict__ in_idx,
                 const int* __restrict__ out_idx,
                 float* __restrict__ acc,
                 int P, int tiles_per_k) {
    int k = blockIdx.y;
    int waveInK = blockIdx.x * (blockDim.x >> 5) + (threadIdx.x >> 5);
    int t0 = waveInK * TPW;
    if (t0 >= tiles_per_k) return;            // uniform per wave -> EXEC all 1s
    int lane = threadIdx.x & 31;
    int row  = lane & 15;                     // matrix row M this lane serves
    int half = lane >> 4;                     // which K-half of the row
    int col  = lane & 15;

    // ---- load packed B tiles once (each 32x16 bf16) ----
    const unsigned short* wb = Wlay + ((long long)(k * 2) * 32 + lane) * 16;
    v16bf b0 = *(const v16bf*)(wb);
    v16bf b1 = *(const v16bf*)(wb + 32 * 16);

    int kbase = k * P;
#pragma unroll
    for (int it = 0; it < TPW; ++it) {
        int t = t0 + it;
        if (t < tiles_per_k) {                // uniform per wave
            int base = kbase + t * 16;
            int ii   = in_idx[base + row];
            int oi   = out_idx[base + row];

            // gather A (16x32 bf16): two contiguous 16B chunks per lane
            const unsigned short* xrow = xb + (long long)ii * CCH;
            v8bf alo = *(const v8bf*)(xrow + half * 8);       // K: half*8..+7
            v8bf ahi = *(const v8bf*)(xrow + 16 + half * 8);  // K: 16+half*8..+7
            v16bf a = __builtin_shufflevector(alo, ahi,
                0, 1, 2, 3, 4, 5, 6, 7, 8, 9, 10, 11, 12, 13, 14, 15);

            // D = A x B, f32 accumulate
            v8f c0 = {};
            v8f c1 = {};
            c0 = __builtin_amdgcn_wmma_f32_16x16x32_bf16(false, a, false, b0,
                                                         (short)0, c0, false, false);
            c1 = __builtin_amdgcn_wmma_f32_16x16x32_bf16(false, a, false, b1,
                                                         (short)0, c1, false, false);

            // scatter-add: lane holds rows M = half*8 + r, column = col
#pragma unroll
            for (int r = 0; r < 8; ++r) {
                int m  = half * 8 + r;
                int om = __shfl(oi, m, 32);   // output row index for M=m
                float* dst = acc + (long long)om * CCH;
                atomicAdd(dst + col,      c0[r]);
                atomicAdd(dst + col + 16, c1[r]);
            }
        }
    }
}

// ---------------------------------------------------------------------------
// Remainder pairs (P % 16 != 0): scalar fallback in f32. One thread per
// (k, pair, out_channel).
// ---------------------------------------------------------------------------
__global__ void sparse_conv_tail(const float* __restrict__ x,
                                 const float* __restrict__ W,
                                 const int* __restrict__ in_idx,
                                 const int* __restrict__ out_idx,
                                 float* __restrict__ acc,
                                 int P, int pstart, int rem, int K) {
    int i = blockIdx.x * blockDim.x + threadIdx.x;
    int total = K * rem * CCH;
    if (i >= total) return;
    int c  = i & (CCH - 1);
    int pr = (i >> 5) % rem;
    int k  = (i >> 5) / rem;
    int p  = pstart + pr;
    int ii = in_idx[k * P + p];
    int oi = out_idx[k * P + p];
    const float* xr = x + (long long)ii * CCH;
    const float* wk = W + (long long)k * CCH * CCH + c;
    float s = 0.0f;
#pragma unroll
    for (int q = 0; q < CCH; ++q) s += xr[q] * wk[q * CCH];
    atomicAdd(&acc[(long long)oi * CCH + c], s);
}

// ---------------------------------------------------------------------------
// Per-channel sum / sumsq over acc [N,32]. One wave reads whole rows
// (coalesced 128B); channel = lane.
// ---------------------------------------------------------------------------
__global__ void __launch_bounds__(256)
bn_stats(const float* __restrict__ acc, float* __restrict__ sums, int Nv) {
    __shared__ float ssum[256];
    __shared__ float ssq[256];
    int c = threadIdx.x & 31;
    float s = 0.0f, q = 0.0f;
    int rowStart  = blockIdx.x * (blockDim.x >> 5) + (threadIdx.x >> 5);
    int rowStride = gridDim.x * (blockDim.x >> 5);
    for (int r = rowStart; r < Nv; r += rowStride) {
        float v = acc[(long long)r * CCH + c];
        s += v;
        q += v * v;
    }
    ssum[threadIdx.x] = s;
    ssq[threadIdx.x]  = q;
    __syncthreads();
    if (threadIdx.x < 32) {
#pragma unroll
        for (int w = 1; w < 8; ++w) {
            s += ssum[threadIdx.x + 32 * w];
            q += ssq[threadIdx.x + 32 * w];
        }
        atomicAdd(&sums[c], s);
        atomicAdd(&sums[32 + c], q);
    }
}

// ---------------------------------------------------------------------------
// Fused batchnorm (batch stats) + ReLU, in place over d_out.
// ---------------------------------------------------------------------------
__global__ void bn_relu_apply(float* __restrict__ y,
                              const float* __restrict__ sums,
                              const float* __restrict__ gamma,
                              const float* __restrict__ beta,
                              long long total, float invN) {
    long long i = (long long)blockIdx.x * blockDim.x + threadIdx.x;
    if (i >= total) return;
    int c = (int)(i & (CCH - 1));
    float mean = sums[c] * invN;
    float var  = sums[32 + c] * invN - mean * mean;
    float sc   = gamma[c] * rsqrtf(var + BN_EPS);
    float bb   = beta[c] - mean * sc;
    float v    = y[i] * sc + bb;
    y[i] = v > 0.0f ? v : 0.0f;
}

// ---------------------------------------------------------------------------
// Launcher
// ---------------------------------------------------------------------------
extern "C" void kernel_launch(void* const* d_in, const int* in_sizes, int n_in,
                              void* d_out, int out_size, void* d_ws, size_t ws_size,
                              hipStream_t stream) {
    const float* x      = (const float*)d_in[0];
    const float* W      = (const float*)d_in[1];
    const float* gamma  = (const float*)d_in[2];
    const float* beta   = (const float*)d_in[3];
    const int*   in_idx = (const int*)d_in[4];
    const int*   out_idx= (const int*)d_in[5];
    float*       out    = (float*)d_out;

    const int N = in_sizes[0] / CCH;                 // voxels
    const int K = in_sizes[1] / (CCH * CCH);         // kernel offsets
    const int P = in_sizes[4] / K;                   // pairs per offset

    // workspace layout (256B aligned regions)
    char* ws = (char*)d_ws;
    size_t xbBytes = (size_t)N * CCH * sizeof(unsigned short);
    xbBytes = (xbBytes + 255) & ~(size_t)255;
    unsigned short* xb   = (unsigned short*)ws;
    size_t wlayBytes = (size_t)K * 2 * 32 * 16 * sizeof(unsigned short);
    wlayBytes = (wlayBytes + 255) & ~(size_t)255;
    unsigned short* Wlay = (unsigned short*)(ws + xbBytes);
    float* sums          = (float*)(ws + xbBytes + wlayBytes);   // 64 floats

    const long long totalElems = (long long)N * CCH;

    // 1) zero accumulator (d_out doubles as f32 accumulator) + stats buffer
    zero_f32<<<2048, 256, 0, stream>>>(out, totalElems);
    zero_f32<<<1, 64, 0, stream>>>(sums, 64);

    // 2) convert x -> bf16 ; pack W -> per-lane WMMA-B layout
    convert_x_bf16<<<2048, 256, 0, stream>>>(x, xb, totalElems / 4);
    int wlayTotal = K * 2 * 32 * 16;
    pack_w_bf16<<<(wlayTotal + 255) / 256, 256, 0, stream>>>(W, Wlay, wlayTotal);

    // 3) WMMA sparse conv: one wave per TPW*16 pairs; k on blockIdx.y
    int tiles_per_k = P / 16;
    if (tiles_per_k > 0) {
        int wavesPerBlock = 256 / 32;
        int tilesPerBlock = wavesPerBlock * TPW;
        dim3 grid((tiles_per_k + tilesPerBlock - 1) / tilesPerBlock, K);
        sparse_conv_wmma<<<grid, 256, 0, stream>>>(
            xb, Wlay, in_idx, out_idx, out, P, tiles_per_k);
    }
    int rem = P - tiles_per_k * 16;
    if (rem > 0) {
        int tailTotal = K * rem * CCH;
        sparse_conv_tail<<<(tailTotal + 255) / 256, 256, 0, stream>>>(
            x, W, in_idx, out_idx, out, P, tiles_per_k * 16, rem, K);
    }

    // 4) batchnorm statistics, then fused normalize + ReLU in place
    bn_stats<<<1024, 256, 0, stream>>>(out, sums, N);
    bn_relu_apply<<<(int)((totalElems + 255) / 256), 256, 0, stream>>>(
        out, sums, gamma, beta, totalElems, 1.0f / (float)N);
}
